// Sage_63239098466920
// MI455X (gfx1250) — compile-verified, matrix operand order checked by reading, not run on hardware
//
#include <hip/hip_runtime.h>
#include <hip/hip_bf16.h>
#include <math.h>

typedef __attribute__((ext_vector_type(16))) _Float16 v16h;
typedef __attribute__((ext_vector_type(8)))  float    v8f;

#define DIM    128
#define HID    64
#define NCLS   40

__device__ __forceinline__ void atomAddF(float* p, float v) {
    // lowers to global_atomic_add_f32 on gfx1250
    unsafeAtomicAdd(p, v);
}

// ---------------------------------------------------------------- zero scratch
__global__ __launch_bounds__(256) void zerof_kernel(float* __restrict__ p, long n) {
    long i = (long)blockIdx.x * blockDim.x + threadIdx.x;
    long stride = (long)gridDim.x * blockDim.x;
    for (; i < n; i += stride) p[i] = 0.0f;
}

// ------------------------------------------------------------- weight packing
// Pre-swizzle a [K x Nw] f32 weight matrix into f16 WMMA B-operand order:
// blob[tile][lane][hi], tile = kt*NT + nt, so a wave fetches one whole B tile
// as a single 32-byte vector load. Columns >= NC are zero-padded.
__global__ __launch_bounds__(256) void pack_kernel(
    const float* __restrict__ W, _Float16* __restrict__ out,
    int NT, int Nw, int NC, int total)
{
    int id = blockIdx.x * 256 + threadIdx.x;
    if (id >= total) return;
    int hi   = id & 15;
    int lane = (id >> 4) & 31;
    int t    = id >> 9;               // tile index = kt*NT + nt
    int nt   = t % NT;
    int kt   = t / NT;
    int lh = lane >> 4, lr = lane & 15;
    // ISA 16-bit B layout (mirror of A): halves 0..7 -> K = kb+0..7,
    // halves 8..15 -> K = kb+16..23, kb = kt*32 + 8*lh
    int k = kt * 32 + 8 * lh + (hi & 7) + ((hi >> 3) << 4);
    int c = nt * 16 + lr;
    float v = (c < NC) ? W[(long)k * Nw + c] : 0.0f;
    out[(long)t * 512 + lane * 16 + hi] = (_Float16)v;
}

// -------------------------------------------------- scatter layer 1 (D = 128)
// one thread handles one float4 chunk of one edge: E * 32 threads
__global__ __launch_bounds__(256) void scatter1_kernel(
    const float* __restrict__ x, const int* __restrict__ src,
    const int* __restrict__ dst, float* __restrict__ agg,
    float* __restrict__ deg, int E)
{
    long gid = (long)blockIdx.x * blockDim.x + threadIdx.x;
    int e = (int)(gid >> 5);
    if (e >= E) return;
    int part = (int)(gid & 31);
    int c = part * 4;
    int s = src[e], d = dst[e];
    const float4 v = *(const float4*)(x + (long)s * DIM + c);
    float* p = agg + (long)d * DIM + c;
    atomAddF(p + 0, v.x);
    atomAddF(p + 1, v.y);
    atomAddF(p + 2, v.z);
    atomAddF(p + 3, v.w);
    if (part == 0) atomAddF(deg + d, 1.0f);
}

// --------------------------------------------------- scatter layer 2 (H = 64)
__global__ __launch_bounds__(256) void scatter2_kernel(
    const float* __restrict__ h, const int* __restrict__ src,
    const int* __restrict__ dst, float* __restrict__ agg, int E)
{
    long gid = (long)blockIdx.x * blockDim.x + threadIdx.x;
    int e = (int)(gid >> 4);
    if (e >= E) return;
    int part = (int)(gid & 15);
    int c = part * 4;
    int s = src[e], d = dst[e];
    const float4 v = *(const float4*)(h + (long)s * HID + c);
    float* p = agg + (long)d * HID + c;
    atomAddF(p + 0, v.x);
    atomAddF(p + 1, v.y);
    atomAddF(p + 2, v.z);
    atomAddF(p + 3, v.w);
}

__device__ __forceinline__ v8f wmma_f16(v16h a, v16h b, v8f c) {
    return __builtin_amdgcn_wmma_f32_16x16x32_f16(
        /*neg_a=*/false, a, /*neg_b=*/false, b,
        /*c_mod=*/(short)0, c, /*reuse_a=*/false, /*reuse_b=*/false);
}

// ------------------------------- layer 1: h = relu(agg/deg @ W1l + x @ W1r + b1)
// one wave computes a 16x64 output tile; 4 waves / block
__global__ __launch_bounds__(128) void gemm1_kernel(
    const float* __restrict__ xsum, const float* __restrict__ x,
    const float* __restrict__ deg,
    const _Float16* __restrict__ pw1l, const _Float16* __restrict__ pw1r,
    const float* __restrict__ b1, float* __restrict__ h, int N)
{
    const int lane = threadIdx.x & 31;
    const int wave = threadIdx.x >> 5;
    const int m0 = (blockIdx.x * 4 + wave) << 4;
    if (m0 >= N) return;                 // wave-uniform, no barriers in kernel
    const int lr = lane & 15;
    const int lh = lane >> 4;

    int r = m0 + lr;
    if (r >= N) r = N - 1;               // clamp loads; stores are guarded
    const float rdeg = 1.0f / fmaxf(deg[r], 1.0f);
    const float* __restrict__ rowA = xsum + (long)r * DIM;
    const float* __restrict__ rowX = x    + (long)r * DIM;

    const v8f vzero = {0.f, 0.f, 0.f, 0.f, 0.f, 0.f, 0.f, 0.f};
    v8f acc[4];
#pragma unroll
    for (int nt = 0; nt < 4; ++nt) acc[nt] = vzero;

#pragma unroll
    for (int kt = 0; kt < 4; ++kt) {
        const int kb = kt * 32 + 8 * lh; // ISA 16-bit A layout: lane-half K base
        v16h Aa, Ax;
#pragma unroll
        for (int i = 0; i < 8; ++i) {
            Aa[i]     = (_Float16)(rowA[kb + i]      * rdeg);
            Aa[8 + i] = (_Float16)(rowA[kb + 16 + i] * rdeg);
            Ax[i]     = (_Float16)(rowX[kb + i]);
            Ax[8 + i] = (_Float16)(rowX[kb + 16 + i]);
        }
#pragma unroll
        for (int nt = 0; nt < 4; ++nt) {
            const v16h Bl = *(const v16h*)(pw1l + ((long)(kt * 4 + nt) * 512) + lane * 16);
            const v16h Br = *(const v16h*)(pw1r + ((long)(kt * 4 + nt) * 512) + lane * 16);
            acc[nt] = wmma_f16(Aa, Bl, acc[nt]);
            acc[nt] = wmma_f16(Ax, Br, acc[nt]);
        }
    }

#pragma unroll
    for (int nt = 0; nt < 4; ++nt) {
        const int c = nt * 16 + lr;
        const float bias = b1[c];
#pragma unroll
        for (int j = 0; j < 8; ++j) {
            const int m = m0 + 8 * lh + j;   // C/D layout: vgpr j, lane-half
            if (m < N)
                h[(long)m * HID + c] = fmaxf(acc[nt][j] + bias, 0.0f);
        }
    }
}

// ---- layer 2: out = log_softmax(agg2/deg @ W2l + h @ W2r + b2), C=40 (pad 48)
__global__ __launch_bounds__(128) void gemm2_kernel(
    const float* __restrict__ hsum, const float* __restrict__ h,
    const float* __restrict__ deg,
    const _Float16* __restrict__ pw2l, const _Float16* __restrict__ pw2r,
    const float* __restrict__ b2, float* __restrict__ out, int N)
{
    __shared__ float lds[4][16][49];     // padded stride to dodge bank conflicts

    const int lane = threadIdx.x & 31;
    const int wave = threadIdx.x >> 5;
    const int m0 = (blockIdx.x * 4 + wave) << 4;
    const int lr = lane & 15;
    const int lh = lane >> 4;

    int r = m0 + lr;
    if (r >= N) r = N - 1;
    const float rdeg = 1.0f / fmaxf(deg[r], 1.0f);
    const float* __restrict__ rowA = hsum + (long)r * HID;
    const float* __restrict__ rowH = h    + (long)r * HID;

    const v8f vzero = {0.f, 0.f, 0.f, 0.f, 0.f, 0.f, 0.f, 0.f};
    v8f acc[3];
#pragma unroll
    for (int nt = 0; nt < 3; ++nt) acc[nt] = vzero;

#pragma unroll
    for (int kt = 0; kt < 2; ++kt) {
        const int kb = kt * 32 + 8 * lh;
        v16h Aa, Ah;
#pragma unroll
        for (int i = 0; i < 8; ++i) {
            Aa[i]     = (_Float16)(rowA[kb + i]      * rdeg);
            Aa[8 + i] = (_Float16)(rowA[kb + 16 + i] * rdeg);
            Ah[i]     = (_Float16)(rowH[kb + i]);
            Ah[8 + i] = (_Float16)(rowH[kb + 16 + i]);
        }
#pragma unroll
        for (int nt = 0; nt < 3; ++nt) {
            const v16h Bl = *(const v16h*)(pw2l + ((long)(kt * 3 + nt) * 512) + lane * 16);
            const v16h Br = *(const v16h*)(pw2r + ((long)(kt * 3 + nt) * 512) + lane * 16);
            acc[nt] = wmma_f16(Aa, Bl, acc[nt]);
            acc[nt] = wmma_f16(Ah, Br, acc[nt]);
        }
    }

    // stage logits (pre-softmax) into LDS
#pragma unroll
    for (int nt = 0; nt < 3; ++nt) {
        const int c = nt * 16 + lr;
        const float bias = (c < NCLS) ? b2[c] : 0.0f;
#pragma unroll
        for (int j = 0; j < 8; ++j)
            lds[wave][8 * lh + j][c] = acc[nt][j] + bias;
    }
    __syncthreads();

    // lanes 0..15 of each wave: one row each, log_softmax over 40 classes
    if (lh == 0) {
        const int row = m0 + lr;
        if (row < N) {
            const float* rp = &lds[wave][lr][0];
            float mx = rp[0];
#pragma unroll
            for (int c = 1; c < NCLS; ++c) mx = fmaxf(mx, rp[c]);
            float s = 0.0f;
#pragma unroll
            for (int c = 0; c < NCLS; ++c) s += expf(rp[c] - mx);
            const float lse = mx + logf(s);
            float* op = out + (long)row * NCLS;
#pragma unroll
            for (int c = 0; c < NCLS; ++c) op[c] = rp[c] - lse;
        }
    }
}

extern "C" void kernel_launch(void* const* d_in, const int* in_sizes, int n_in,
                              void* d_out, int out_size, void* d_ws, size_t ws_size,
                              hipStream_t stream) {
    const float* x   = (const float*)d_in[0];
    const int*   ei  = (const int*)d_in[1];
    const float* W1l = (const float*)d_in[2];
    const float* W1r = (const float*)d_in[3];
    const float* b1  = (const float*)d_in[4];
    const float* W2l = (const float*)d_in[5];
    const float* W2r = (const float*)d_in[6];
    const float* b2  = (const float*)d_in[7];
    float* out = (float*)d_out;

    const int N = in_sizes[0] / DIM;       // 100000
    const int E = in_sizes[1] / 2;         // 1600000
    const int* src = ei;
    const int* dst = ei + E;

    // workspace layout (floats): agg1[N*128] | deg[N] | agg2[N*64] | h[N*64]
    // then (halves): pw1l[8192] | pw1r[8192] | pw2l[3072] | pw2r[3072]
    float* agg1 = (float*)d_ws;
    float* deg  = agg1 + (long)N * DIM;
    float* agg2 = deg + N;
    float* h    = agg2 + (long)N * HID;
    _Float16* pw1l = (_Float16*)(h + (long)N * HID);
    _Float16* pw1r = pw1l + 8192;
    _Float16* pw2l = pw1r + 8192;
    _Float16* pw2r = pw2l + 3072;

    const long zcount = (long)N * DIM + N + (long)N * HID;
    zerof_kernel<<<2048, 256, 0, stream>>>(agg1, zcount);

    // pack weights into WMMA-B-operand order (f16, zero-padded columns)
    pack_kernel<<<(8192 + 255) / 256, 256, 0, stream>>>(W1l, pw1l, 4, HID,  HID,  8192);
    pack_kernel<<<(8192 + 255) / 256, 256, 0, stream>>>(W1r, pw1r, 4, HID,  HID,  8192);
    pack_kernel<<<(3072 + 255) / 256, 256, 0, stream>>>(W2l, pw2l, 3, NCLS, NCLS, 3072);
    pack_kernel<<<(3072 + 255) / 256, 256, 0, stream>>>(W2r, pw2r, 3, NCLS, NCLS, 3072);

    {   // scatter x into agg1 + degree count
        long threads = (long)E * 32;
        int grid = (int)((threads + 255) / 256);
        scatter1_kernel<<<grid, 256, 0, stream>>>(x, src, dst, agg1, deg, E);
    }

    {   // layer-1 fused GEMM (mean-scale + bias + relu)
        int grid = (N + 63) / 64;
        gemm1_kernel<<<grid, 128, 0, stream>>>(agg1, x, deg, pw1l, pw1r, b1, h, N);
    }

    {   // scatter h into agg2
        long threads = (long)E * 16;
        int grid = (int)((threads + 255) / 256);
        scatter2_kernel<<<grid, 256, 0, stream>>>(h, src, dst, agg2, E);
    }

    {   // layer-2 fused GEMM + log_softmax
        int grid = (N + 63) / 64;
        gemm2_kernel<<<grid, 128, 0, stream>>>(agg2, h, deg, pw2l, pw2r, b2, out, N);
    }
}
